// SparseCoder_4346506904136
// MI455X (gfx1250) — matrix operand order repaired and checked by reading, not compiled, not run on hardware
//
#include <hip/hip_runtime.h>
#include <math.h>

// ---------------------------------------------------------------------------
// FISTA sparse coding on gfx1250 (MI455X), split-bf16 WMMA GEMM core.
// All GEMM operands are pre-split (f32 -> bf16 hi/lo) and pre-swizzled into
// WMMA fragment order so the GEMM mainloop is pure load+WMMA.
// ---------------------------------------------------------------------------

typedef __attribute__((ext_vector_type(16))) __bf16 v16bf;
typedef __attribute__((ext_vector_type(8)))  float  v8f;

#define BATCH_  8192
#define KBAS_   2048
#define DIM_    1024
#define LAM_    0.1f

// ---- f32 <-> bf16 helpers (RNE) -------------------------------------------
__device__ __forceinline__ unsigned short f2bf(float v) {
  unsigned u = __float_as_uint(v);
  unsigned r = u + 0x7FFFu + ((u >> 16) & 1u);
  return (unsigned short)(r >> 16);
}
__device__ __forceinline__ float bf2f(unsigned short b) {
  return __uint_as_float(((unsigned)b) << 16);
}

// ---- pack B operand (phi / phi^T) into WMMA B-fragment order, hi/lo -------
// B-frag layout (32x16 bf16): lane: n = lane&15 (col), element e -> k = (lane>>4)*16 + e
// packed[((nt*KT + kt)*32 + lane)*16 + e] = B[kt*32 + (lane>>4)*16 + e][nt*16 + (lane&15)]
__global__ __launch_bounds__(256)
void packB_kernel(const float* __restrict__ src, int pitch, int trans, int KT,
                  unsigned short* __restrict__ dh, unsigned short* __restrict__ dl,
                  int total) {
  int gid = blockIdx.x * 256 + threadIdx.x;
  if (gid >= total) return;
  int e    = gid & 15;
  int lane = (gid >> 4) & 31;
  int t    = gid >> 9;
  int kt = t % KT, nt = t / KT;
  int k = kt * 32 + (lane >> 4) * 16 + e;
  int n = nt * 16 + (lane & 15);
  float v = trans ? src[(size_t)n * pitch + k] : src[(size_t)k * pitch + n];
  unsigned short hb = f2bf(v);
  dh[gid] = hb;
  dl[gid] = f2bf(v - bf2f(hb));
}

// ---- pack A operand into WMMA A-fragment order, hi/lo ---------------------
// A-frag layout (16x32 bf16): lane: m = lane&15 (row), half = lane>>4,
// element e -> k = half*16 + e
// packed[((mt*KT + kt)*32 + lane)*16 + e] = A[mt*16 + (lane&15)][kt*32 + kidx]
__global__ __launch_bounds__(256)
void packA_kernel(const float* __restrict__ src, int ldk, int KT,
                  unsigned short* __restrict__ dh, unsigned short* __restrict__ dl,
                  int total) {
  int gid = blockIdx.x * 256 + threadIdx.x;
  if (gid >= total) return;
  int e    = gid & 15;
  int lane = (gid >> 4) & 31;
  int t    = gid >> 9;
  int kt = t % KT, mt = t / KT;
  int half = lane >> 4;
  int kidx = half * 16 + e;
  int row = mt * 16 + (lane & 15);
  float v = src[(size_t)row * ldk + kt * 32 + kidx];
  unsigned short hb = f2bf(v);
  dh[gid] = hb;
  dl[gid] = f2bf(v - bf2f(hb));
}

// ---- main split-bf16 WMMA GEMM, 64x64 tile per wave -----------------------
// C = A*B with A,B pre-split bf16 hi/lo in fragment order; f32 accumulate;
// fp32 emulated via hi*hi + lo*hi + hi*lo.
// MODE 0: Out = C              (G = phi*phi^T, recon = alpha*phi)
// MODE 1: Out = C - X          (r = y*phi - x)
// MODE 2: fused FISTA update   (grad = r*phi^T; prox + momentum, in-place)
template <int MODE>
__global__ __launch_bounds__(256)
void gemm64(const unsigned short* __restrict__ Ah,
            const unsigned short* __restrict__ Al,
            const unsigned short* __restrict__ Bh,
            const unsigned short* __restrict__ Bl,
            int N, int KT,
            float* __restrict__ Out, const float* __restrict__ X,
            float* __restrict__ Yb, float* __restrict__ Pb,
            const float* __restrict__ sc, float cmom) {
  const int lane = threadIdx.x & 31;
  const int half = lane >> 4;
  const int mr   = lane & 15;
  const int wid  = blockIdx.x * 8 + (threadIdx.x >> 5);
  const int NBT  = N >> 6;
  const int wm = wid / NBT, wn = wid % NBT;
  const int row0 = wm * 64, col0 = wn * 64;

  const unsigned short* ahp[4];
  const unsigned short* alp[4];
#pragma unroll
  for (int rt = 0; rt < 4; ++rt) {
    size_t o = (size_t)(4 * wm + rt) * KT * 512 + (size_t)lane * 16;
    ahp[rt] = Ah + o; alp[rt] = Al + o;
  }
  const unsigned short* bhp[4];
  const unsigned short* blp[4];
#pragma unroll
  for (int ct = 0; ct < 4; ++ct) {
    size_t o = (size_t)(4 * wn + ct) * KT * 512 + (size_t)lane * 16;
    bhp[ct] = Bh + o; blp[ct] = Bl + o;
  }

  const v8f vzero = {0.f, 0.f, 0.f, 0.f, 0.f, 0.f, 0.f, 0.f};
  v8f acc[4][4];
#pragma unroll
  for (int i = 0; i < 4; ++i)
#pragma unroll
    for (int j = 0; j < 4; ++j) acc[i][j] = vzero;

  for (int kt = 0; kt < KT; ++kt) {
    const size_t ko = (size_t)kt * 512;
    v16bf ah[4], al[4];
#pragma unroll
    for (int rt = 0; rt < 4; ++rt) {
      ah[rt] = *reinterpret_cast<const v16bf*>(ahp[rt] + ko);
      al[rt] = *reinterpret_cast<const v16bf*>(alp[rt] + ko);
    }
#pragma unroll
    for (int ct = 0; ct < 4; ++ct) {
      v16bf bh = *reinterpret_cast<const v16bf*>(bhp[ct] + ko);
      v16bf bl = *reinterpret_cast<const v16bf*>(blp[ct] + ko);
#pragma unroll
      for (int rt = 0; rt < 4; ++rt) {
        v8f c = acc[rt][ct];
        c = __builtin_amdgcn_wmma_f32_16x16x32_bf16(false, ah[rt], false, bh,
                                                    (short)0, c, false, false);
        c = __builtin_amdgcn_wmma_f32_16x16x32_bf16(false, al[rt], false, bh,
                                                    (short)0, c, false, false);
        c = __builtin_amdgcn_wmma_f32_16x16x32_bf16(false, ah[rt], false, bl,
                                                    (short)0, c, false, false);
        acc[rt][ct] = c;
      }
    }
  }

  float step = 0.f, thr = 0.f;
  if constexpr (MODE == 2) { step = sc[0]; thr = sc[1]; }

#pragma unroll
  for (int rt = 0; rt < 4; ++rt) {
#pragma unroll
    for (int ct = 0; ct < 4; ++ct) {
      const int col = col0 + ct * 16 + mr;
#pragma unroll
      for (int rr = 0; rr < 8; ++rr) {
        const int row = row0 + rt * 16 + half * 8 + rr;
        const size_t idx = (size_t)row * N + col;
        float val = acc[rt][ct][rr];
        if constexpr (MODE == 0) {
          Out[idx] = val;
        } else if constexpr (MODE == 1) {
          Out[idx] = val - X[idx];
        } else {
          float y = Yb[idx];
          float z = fmaf(-step, val, y);
          float az = fabsf(z) - thr;
          float a = az > 0.f ? copysignf(az, z) : 0.f;
          float p = Pb[idx];
          Pb[idx] = a;
          Yb[idx] = fmaf(cmom, a - p, a);
        }
      }
    }
  }
}

// ---- power iteration helpers (deterministic, no float atomics) ------------
__global__ void powinit(float* __restrict__ v) {
  int j = blockIdx.x * 256 + threadIdx.x;
  v[j] = 1.f;
}

__global__ __launch_bounds__(256)
void matvec2048(const float* __restrict__ G, const float* __restrict__ v,
                float* __restrict__ w, float* __restrict__ pSq,
                float* __restrict__ pRay) {
  __shared__ float red[256];
  const int tid = threadIdx.x;
  const int j = blockIdx.x * 256 + tid;
  const float4* g4 = reinterpret_cast<const float4*>(G + (size_t)j * 2048);
  const float4* v4 = reinterpret_cast<const float4*>(v);
  float s = 0.f;
  for (int i = 0; i < 512; ++i) {
    float4 a = g4[i], b = v4[i];
    s = fmaf(a.x, b.x, s); s = fmaf(a.y, b.y, s);
    s = fmaf(a.z, b.z, s); s = fmaf(a.w, b.w, s);
  }
  w[j] = s;
  red[tid] = s * s; __syncthreads();
  for (int st = 128; st > 0; st >>= 1) {
    if (tid < st) red[tid] += red[tid + st];
    __syncthreads();
  }
  if (tid == 0) pSq[blockIdx.x] = red[0];
  __syncthreads();
  red[tid] = s * v[j]; __syncthreads();
  for (int st = 128; st > 0; st >>= 1) {
    if (tid < st) red[tid] += red[tid + st];
    __syncthreads();
  }
  if (tid == 0) pRay[blockIdx.x] = red[0];
}

__global__ void pnorm(float* __restrict__ v, const float* __restrict__ w,
                      const float* __restrict__ pSq) {
  float ss = 0.f;
#pragma unroll
  for (int i = 0; i < 8; ++i) ss += pSq[i];
  int j = blockIdx.x * 256 + threadIdx.x;
  v[j] = w[j] / sqrtf(ss);
}

__global__ void fin_scalars(float* __restrict__ sc, const float* __restrict__ pRay) {
  float L = 0.f;
#pragma unroll
  for (int i = 0; i < 8; ++i) L += pRay[i];
  float step = 1.f / L;
  sc[0] = step;
  sc[1] = LAM_ * step;
}

__global__ void zeroK(float4* __restrict__ p, int n4) {
  int i = blockIdx.x * 256 + threadIdx.x;
  if (i < n4) p[i] = make_float4(0.f, 0.f, 0.f, 0.f);
}

// ---------------------------------------------------------------------------
extern "C" void kernel_launch(void* const* d_in, const int* in_sizes, int n_in,
                              void* d_out, int out_size, void* d_ws, size_t ws_size,
                              hipStream_t stream) {
  (void)in_sizes; (void)n_in; (void)out_size; (void)ws_size;
  const float* x   = (const float*)d_in[0];  // [8192, 1024]
  const float* phi = (const float*)d_in[1];  // [2048, 1024]
  float* out = (float*)d_out;                // alpha [8192,2048] ++ recon [8192,1024]

  const int Bn = BATCH_, K = KBAS_, D = DIM_;

  // workspace layout
  char* w8 = (char*)d_ws;
  float* hdr = (float*)w8;
  float* sc   = hdr;        // [0]=step, [1]=thr
  float* pSq  = hdr + 8;    // 8 block partials
  float* pRay = hdr + 16;   // 8 block partials
  float* pv   = hdr + 2048; // v[2048]
  float* pw   = hdr + 4096; // w[2048]
  size_t off = 32768;
  unsigned short* phiP_h = (unsigned short*)(w8 + off); off += (size_t)K * D * 2;
  unsigned short* phiP_l = (unsigned short*)(w8 + off); off += (size_t)K * D * 2;
  unsigned short* phiT_h = (unsigned short*)(w8 + off); off += (size_t)K * D * 2;
  unsigned short* phiT_l = (unsigned short*)(w8 + off); off += (size_t)K * D * 2;
  float* Y = (float*)(w8 + off); off += (size_t)Bn * K * 4;   // 64MB
  float* P = (float*)(w8 + off); off += (size_t)Bn * K * 4;   // 64MB
  float* R = (float*)(w8 + off); off += (size_t)Bn * D * 4;   // 32MB
  float* G = R;  // G (16MB) is dead before R (32MB) is first written
  unsigned short* Ya_h = (unsigned short*)(w8 + off); off += (size_t)Bn * K * 2; // 32MB
  unsigned short* Ya_l = (unsigned short*)(w8 + off); off += (size_t)Bn * K * 2; // 32MB
  unsigned short* Ra_h = (unsigned short*)(w8 + off); off += (size_t)Bn * D * 2; // 16MB
  unsigned short* Ra_l = (unsigned short*)(w8 + off); off += (size_t)Bn * D * 2; // 16MB
  // reuse: phi-as-A pack (2048x1024 -> 4MB x2) lives in Ra_*;
  //        alpha-as-A pack (8192x2048) lives in Ya_* after the FISTA loop.
  unsigned short* phiA_h = Ra_h;
  unsigned short* phiA_l = Ra_l;
  unsigned short* Pa_h = Ya_h;
  unsigned short* Pa_l = Ya_l;

  // zero init alpha/y state (deterministic across replays)
  {
    int n4 = (Bn * K) / 4;
    zeroK<<<(n4 + 255) / 256, 256, 0, stream>>>((float4*)Y, n4);
    zeroK<<<(n4 + 255) / 256, 256, 0, stream>>>((float4*)P, n4);
  }

  // pack phi as B (inner=K) and phi^T as B (B[d][j]=phi[j][d], inner=D)
  {
    int total = K * D;
    packB_kernel<<<total / 256, 256, 0, stream>>>(phi, D, 0, K / 32, phiP_h, phiP_l, total);
    packB_kernel<<<total / 256, 256, 0, stream>>>(phi, D, 1, D / 32, phiT_h, phiT_l, total);
  }

  // G = phi @ phi^T : M=2048, N=2048, inner=1024
  {
    int total = K * D;
    packA_kernel<<<total / 256, 256, 0, stream>>>(phi, D, D / 32, phiA_h, phiA_l, total);
    gemm64<0><<<(K / 64) * (K / 64) / 8, 256, 0, stream>>>(
        phiA_h, phiA_l, phiT_h, phiT_l, K, D / 32, G, nullptr, nullptr, nullptr,
        nullptr, 0.f);
  }

  // power iteration for L = lambda_max(G)
  powinit<<<8, 256, 0, stream>>>(pv);
  for (int i = 0; i < 50; ++i) {
    matvec2048<<<8, 256, 0, stream>>>(G, pv, pw, pSq, pRay);
    pnorm<<<8, 256, 0, stream>>>(pv, pw, pSq);
  }
  matvec2048<<<8, 256, 0, stream>>>(G, pv, pw, pSq, pRay);  // Rayleigh v'Gv
  fin_scalars<<<1, 1, 0, stream>>>(sc, pRay);

  // FISTA momentum coefficients (data independent -> host)
  float tcur = 1.f, cmom[100];
  for (int i = 0; i < 100; ++i) {
    float tn = 0.5f * (1.f + sqrtf(1.f + 4.f * tcur * tcur));
    cmom[i] = (tcur - 1.f) / tn;
    tcur = tn;
  }

  const int totYK = Bn * K;  // 16M elements
  const int totBD = Bn * D;  // 8M elements

  // FISTA main loop
  for (int i = 0; i < 100; ++i) {
    // pack Y as A (inner=K), then r = Y @ phi - x : M=8192, N=1024, inner=2048
    packA_kernel<<<totYK / 256, 256, 0, stream>>>(Y, K, K / 32, Ya_h, Ya_l, totYK);
    gemm64<1><<<(Bn / 64) * (D / 64) / 8, 256, 0, stream>>>(
        Ya_h, Ya_l, phiP_h, phiP_l, D, K / 32, R, x, nullptr, nullptr, nullptr, 0.f);
    // pack r as A (inner=D), then grad = r @ phi^T fused with prox+momentum
    packA_kernel<<<totBD / 256, 256, 0, stream>>>(R, D, D / 32, Ra_h, Ra_l, totBD);
    gemm64<2><<<(Bn / 64) * (K / 64) / 8, 256, 0, stream>>>(
        Ra_h, Ra_l, phiT_h, phiT_l, K, D / 32, nullptr, nullptr, Y, P, sc, cmom[i]);
  }

  // recon = alpha @ phi -> out[B*K ..]   (alpha = P; pack reuses Ya_* space)
  packA_kernel<<<totYK / 256, 256, 0, stream>>>(P, K, K / 32, Pa_h, Pa_l, totYK);
  gemm64<0><<<(Bn / 64) * (D / 64) / 8, 256, 0, stream>>>(
      Pa_h, Pa_l, phiP_h, phiP_l, D, K / 32, out + (size_t)Bn * K, nullptr,
      nullptr, nullptr, nullptr, 0.f);

  // alpha -> out[0 .. B*K)
  hipMemcpyAsync(out, P, (size_t)Bn * K * sizeof(float),
                 hipMemcpyDeviceToDevice, stream);
}
